// RNN_8787503087759
// MI455X (gfx1250) — matrix operand order repaired and checked by reading, not compiled, not run on hardware
//
#include <hip/hip_runtime.h>

// ---------------------------------------------------------------------------
// 2-layer LSTM for MI455X (gfx1250), bf16 WMMA with f32 accumulation.
// T=256, B=64, D=H=1024.  z = [x,h] @ W + b ; gates i,j,f,o.
//
// Roofline: weights (32 MB bf16 total) live in the 192 MB L2 after first
// touch; each of the 512 recurrent steps streams 16 MB from L2 only.
// The 4 waves of each WG read identical weight blobs -> WGP$ absorbs the
// M-tile redundancy.  Matrix math on v_wmma_f32_16x16x32_bf16.
// ---------------------------------------------------------------------------

typedef __attribute__((ext_vector_type(16))) __bf16 v16bf;
typedef __attribute__((ext_vector_type(8)))  __bf16 v8bf;
typedef __attribute__((ext_vector_type(8)))  float  v8f;

#define T_STEPS 256
#define BATCH   64
#define DEPTH   1024              // D
#define HID     1024              // H
#define KTOT    2048              // D + H
#define NGATE   4096              // 4*H
#define KB32    (KTOT / 32)       // 64 K-blocks of 32
#define NT_G    (HID / 16)        // 64 N-tiles of 16 per gate

// ---------------------------------------------------------------------------
// One-time prep kernels
// ---------------------------------------------------------------------------

__global__ void f32_to_bf16_kernel(const float* __restrict__ x,
                                   __bf16* __restrict__ y, long n) {
    long i = (long)blockIdx.x * blockDim.x + threadIdx.x;
    if (i < n) y[i] = (__bf16)x[i];
}

// Repack W (f32 [KTOT, NGATE] row-major) into bf16 blobs matching the WMMA
// B-operand register layout for v_wmma_f32_16x16x32_bf16 (32x16 K x N tile):
//   lane L (0..31), element e (0..15):
//     k = kb*32 + (e < 8 ? e : e + 8) + (L < 16 ? 0 : 8),   n = nt*16 + (L & 15)
// Blob(nt, kb) lives at halfword offset ((nt*KB32 + kb) * 512 + L*16 + e).
__global__ void pack_w_kernel(const float* __restrict__ W,
                              __bf16* __restrict__ P) {
    long idx = (long)blockIdx.x * blockDim.x + threadIdx.x;
    if (idx >= (long)KTOT * NGATE) return;
    int  e    = (int)(idx & 15);
    int  lane = (int)((idx >> 4) & 31);
    long blob = idx >> 9;
    int  kb   = (int)(blob % KB32);
    int  nt   = (int)(blob / KB32);
    int  k    = kb * 32 + (e < 8 ? e : e + 8) + (lane < 16 ? 0 : 8);
    int  n    = nt * 16 + (lane & 15);
    P[idx] = (__bf16)W[(long)k * NGATE + n];
}

__global__ void zero_u32_kernel(unsigned* __restrict__ p, long n) {
    long i = (long)blockIdx.x * blockDim.x + threadIdx.x;
    if (i < n) p[i] = 0u;
}

// ---------------------------------------------------------------------------
// Fused LSTM step: z = [x,h]@W + b, gate math, c/h update.
// Grid: 128 WGs of 128 threads.  Blocks 0..63 run unit A, 64..127 unit B
// (layer0 step t+1 and layer1 step t are independent).
// Within a unit: blockIdx covers one 16-col N-tile per gate, wave w handles
// M-tile w (rows 16w..16w+15).
// ---------------------------------------------------------------------------

struct StepArgs {
    const __bf16* xpart;    // [BATCH, 1024] bf16 (x_t or lower-layer h_t)
    const __bf16* hpart;    // [BATCH, 1024] bf16 (this layer's h_{t-1})
    const __bf16* Wp;       // packed weights, covers K=2048 x N=4096
    const float*  bias;     // [4096]
    float*        c_state;  // [BATCH, HID] f32 (read old, write new)
    __bf16*       h_state;  // [BATCH, HID] bf16 (write h_t; other parity buf)
    float*        out_f32;  // nullable: [BATCH, HID] slice of d_out
    int           valid;
};

__device__ __forceinline__ float fsig(float x) {
    return __builtin_amdgcn_rcpf(1.0f + __expf(-x));
}
__device__ __forceinline__ float ftanh(float x) {
    // tanh(x) = 1 - 2/(exp(2x)+1); exact limits at +/-inf under rcp.
    return 1.0f - 2.0f * __builtin_amdgcn_rcpf(1.0f + __expf(2.0f * x));
}

// One K-half (32 k-blocks of 32) of the GEMM, clean induction variables only.
//   asrc  : lane-resolved A base  = src + rowA*1024 + abase
//   wbase : lane-resolved B base  = Wp + ((j*KB32 + kbHalf) << 9) + lane*16
__device__ __forceinline__ void gemm_khalf(const __bf16* __restrict__ asrc,
                                           const __bf16* __restrict__ wbase,
                                           v8f acc[4]) {
    #pragma unroll 2
    for (int kb = 0; kb < 32; ++kb) {
        // A tile 16x32 bf16: lane<16 holds K {0..7,16..23}, lane>=16 holds
        // K {8..15,24..31} of row (lane&15)  ->  two contiguous 16B loads.
        union { v16bf v; v8bf h[2]; } A;
        A.h[0] = *(const v8bf*)(asrc + kb * 32);
        A.h[1] = *(const v8bf*)(asrc + kb * 32 + 16);

        // Near-scope prefetch of an upcoming gate-0 weight blob.
        if (kb + 4 < 32) {
            __builtin_prefetch(wbase + ((size_t)(kb + 4) << 9), 0, 3);
        }

        #pragma unroll
        for (int g = 0; g < 4; ++g) {
            const v16bf B = *(const v16bf*)(
                wbase + (((size_t)g * NT_G * KB32 + kb) << 9));
            acc[g] = __builtin_amdgcn_wmma_f32_16x16x32_bf16(
                false, A.v, false, B, (short)0, acc[g], false, false);
        }
    }
}

__global__ __launch_bounds__(128)
void lstm_step2_kernel(StepArgs a0, StepArgs a1) {
    const bool first = (blockIdx.x < 64);
    const __bf16* xpart   = first ? a0.xpart   : a1.xpart;
    const __bf16* hpart   = first ? a0.hpart   : a1.hpart;
    const __bf16* Wp      = first ? a0.Wp      : a1.Wp;
    const float*  bias    = first ? a0.bias    : a1.bias;
    float*        c_state = first ? a0.c_state : a1.c_state;
    __bf16*       h_state = first ? a0.h_state : a1.h_state;
    float*        out_f32 = first ? a0.out_f32 : a1.out_f32;
    const int     valid   = first ? a0.valid   : a1.valid;
    if (!valid) return;                       // block-uniform

    const int lane  = threadIdx.x & 31;
    const int wv    = threadIdx.x >> 5;       // 0..3  -> M-tile
    const int j     = blockIdx.x & 63;        // N-tile within each gate
    const int m0    = wv * 16;
    const int rowA  = m0 + (lane & 15);       // A operand: lane holds row m
    const int abase = (lane < 16) ? 0 : 8;    // A operand K sub-offset

    v8f acc[4];
    #pragma unroll
    for (int g = 0; g < 4; ++g) {
        #pragma unroll
        for (int r = 0; r < 8; ++r) acc[g][r] = 0.0f;
    }

    // K = 2048 split into the x half (k 0..1023) and the h half (k 1024..2047).
    const size_t laneOff = (size_t)rowA * 1024 + abase;
    const __bf16* wb0 = Wp + (((size_t)j * KB32) << 9) + (size_t)lane * 16;
    gemm_khalf(xpart + laneOff, wb0, acc);
    gemm_khalf(hpart + laneOff, wb0 + ((size_t)32 << 9), acc);

    // C/D layout: lane L, VGPR r -> n = j*16 + (L&15),
    //                              m = m0 + r + (L<16 ? 0 : 8)
    const int col = j * 16 + (lane & 15);
    const int mB  = m0 + abase;
    const float bi = bias[0 * HID + col];
    const float bj = bias[1 * HID + col];
    const float bf = bias[2 * HID + col];
    const float bo = bias[3 * HID + col];

    #pragma unroll
    for (int r = 0; r < 8; ++r) {
        const int  m   = mB + r;
        const long off = (long)m * HID + col;
        const float zi = acc[0][r] + bi;
        const float zj = acc[1][r] + bj;
        const float zf = acc[2][r] + bf;
        const float zo = acc[3][r] + bo;
        const float c_old = c_state[off];
        const float ig = fsig(zi);
        const float fg = fsig(zf + 1.0f);     // FORGET_BIAS = 1.0
        const float og = fsig(zo);
        const float jt = ftanh(zj);
        const float c_new = c_old * fg + ig * jt;
        const float h_new = ftanh(c_new) * og;
        c_state[off] = c_new;
        h_state[off] = (__bf16)h_new;
        if (out_f32) out_f32[off] = h_new;
    }
}

// ---------------------------------------------------------------------------
// Host launcher
// ---------------------------------------------------------------------------

extern "C" void kernel_launch(void* const* d_in, const int* in_sizes, int n_in,
                              void* d_out, int out_size, void* d_ws, size_t ws_size,
                              hipStream_t stream) {
    (void)in_sizes; (void)n_in; (void)out_size; (void)ws_size;

    const float* x  = (const float*)d_in[0];   // [256,64,1024]
    const float* W0 = (const float*)d_in[1];   // [2048,4096]
    const float* b0 = (const float*)d_in[2];   // [4096]
    const float* W1 = (const float*)d_in[3];   // [2048,4096]
    const float* b1 = (const float*)d_in[4];   // [4096]
    float* out = (float*)d_out;                // [256,64,1024]

    // Workspace layout (~65 MB):
    char* p = (char*)d_ws;
    __bf16* xb  = (__bf16*)p; p += (size_t)T_STEPS * BATCH * DEPTH * 2;  // 32 MB
    __bf16* W0p = (__bf16*)p; p += (size_t)KTOT * NGATE * 2;             // 16 MB
    __bf16* W1p = (__bf16*)p; p += (size_t)KTOT * NGATE * 2;             // 16 MB
    unsigned* stateBase = (unsigned*)p;
    __bf16* h0b[2]; h0b[0] = (__bf16*)p; p += (size_t)BATCH * HID * 2;
                    h0b[1] = (__bf16*)p; p += (size_t)BATCH * HID * 2;
    __bf16* h1b[2]; h1b[0] = (__bf16*)p; p += (size_t)BATCH * HID * 2;
                    h1b[1] = (__bf16*)p; p += (size_t)BATCH * HID * 2;
    float* c0 = (float*)p; p += (size_t)BATCH * HID * 4;
    float* c1 = (float*)p; p += (size_t)BATCH * HID * 4;

    // 1) Convert x to bf16, repack weights, zero state (1 MB contiguous).
    {
        long nx = (long)T_STEPS * BATCH * DEPTH;
        f32_to_bf16_kernel<<<(nx + 255) / 256, 256, 0, stream>>>(x, xb, nx);
        long nw = (long)KTOT * NGATE;
        pack_w_kernel<<<(nw + 255) / 256, 256, 0, stream>>>(W0, W0p);
        pack_w_kernel<<<(nw + 255) / 256, 256, 0, stream>>>(W1, W1p);
        long nz = ((size_t)BATCH * HID * 2 * 4 /*h bufs*/ +
                   (size_t)BATCH * HID * 4 * 2 /*c bufs*/) / 4;
        zero_u32_kernel<<<(nz + 255) / 256, 256, 0, stream>>>(stateBase, nz);
    }

    // 2) Recurrent steps.  Each launch runs L0(t) and L1(t-1) concurrently
    //    (they are independent); a trailing launch finishes L1(T-1).
    //    Parity: a layer at step s reads h[s&1], writes h[(s+1)&1].
    for (int t = 0; t < T_STEPS; ++t) {
        StepArgs a0;
        a0.xpart   = xb + (size_t)t * BATCH * DEPTH;
        a0.hpart   = h0b[t & 1];
        a0.Wp      = W0p;
        a0.bias    = b0;
        a0.c_state = c0;
        a0.h_state = h0b[(t + 1) & 1];
        a0.out_f32 = nullptr;
        a0.valid   = 1;

        StepArgs a1 = {};
        a1.valid = 0;
        if (t > 0) {
            const int s = t - 1;
            a1.xpart   = h0b[(s + 1) & 1];   // == h0b[t&1]: read-only here
            a1.hpart   = h1b[s & 1];
            a1.Wp      = W1p;
            a1.bias    = b1;
            a1.c_state = c1;
            a1.h_state = h1b[(s + 1) & 1];
            a1.out_f32 = out + (size_t)s * BATCH * HID;
            a1.valid   = 1;
        }
        lstm_step2_kernel<<<128, 128, 0, stream>>>(a0, a1);
    }
    {
        const int s = T_STEPS - 1;
        StepArgs a0 = {}; a0.valid = 0;
        StepArgs a1;
        a1.xpart   = h0b[(s + 1) & 1];
        a1.hpart   = h1b[s & 1];
        a1.Wp      = W1p;
        a1.bias    = b1;
        a1.c_state = c1;
        a1.h_state = h1b[(s + 1) & 1];
        a1.out_f32 = out + (size_t)s * BATCH * HID;
        a1.valid   = 1;
        lstm_step2_kernel<<<128, 128, 0, stream>>>(a0, a1);
    }
}
